// GrammarAttention_24885040513454
// MI455X (gfx1250) — matrix operand order repaired and checked by reading, not compiled, not run on hardware
//
#include <hip/hip_runtime.h>
#include <hip/hip_bf16.h>

typedef __attribute__((ext_vector_type(16))) _Float16 v16h;
typedef __attribute__((ext_vector_type(8)))  _Float16 v8h;
typedef __attribute__((ext_vector_type(8)))  float    v8f;

#define DM   512
#define NH   8
#define DH   64
#define NTOK 1024
#define NB   8
#define NBINS 32

// Build a v16h A/B WMMA fragment from two 16-byte contiguous runs.
__device__ __forceinline__ v16h frag2(const _Float16* p0, const _Float16* p1) {
  v8h a = *(const v8h*)p0;
  v8h b = *(const v8h*)p1;
  v16h r;
#pragma unroll
  for (int i = 0; i < 8; ++i) { r[i] = a[i]; r[i + 8] = b[i]; }
  return r;
}

__device__ __forceinline__ v8f wmma_f16(v16h a, v16h b, v8f c) {
  return __builtin_amdgcn_wmma_f32_16x16x32_f16(false, a, false, b, (short)0, c, false, false);
}

// ---------------- kernel 0: f32 -> f16 convert ----------------
__global__ void k_cvt(const float* __restrict__ src, _Float16* __restrict__ dst, int n) {
  int i = blockIdx.x * blockDim.x + threadIdx.x;
  int stride = gridDim.x * blockDim.x;
  for (; i < n; i += stride) dst[i] = (_Float16)src[i];
}

// ---------------- kernel 1: fused QKV projection (WMMA, 2x2 reg-blocked) ----
// wave tile id t: [0,4096) Q, [4096,8192) K, [8192,12288) V; 32x32 output/wave.
// Q,K: Y[row,feat] = x[row,:] . W[feat,:]   (A = x rows, B = W rows as columns)
// V  : Vt[b][feat][tok] = vw[feat,:] . x[b,tok,:]  (stores V transposed)
__global__ void k_qkv(const _Float16* __restrict__ xh,
                      const _Float16* __restrict__ qwh, const float* __restrict__ qb,
                      const _Float16* __restrict__ kwh, const float* __restrict__ kb,
                      const _Float16* __restrict__ vwh, const float* __restrict__ vb,
                      _Float16* __restrict__ Qh, _Float16* __restrict__ Kh,
                      _Float16* __restrict__ Vth) {
  const int lane = threadIdx.x & 31;
  const int wv   = threadIdx.x >> 5;
  const int t    = blockIdx.x * 4 + wv;
  const int mat  = t >> 12;        // 0=Q 1=K 2=V (4096 tiles each)
  const int t2   = t & 4095;
  const int n16  = lane & 15;
  const int h16  = lane >> 4;

  v8f c00 = {}, c01 = {}, c10 = {}, c11 = {};

  if (mat < 2) {
    const _Float16* W    = (mat == 0) ? qwh : kwh;
    const float*    bias = (mat == 0) ? qb : kb;
    const int tn = t2 & 15, tm = t2 >> 4;       // 16 n-tiles x 256 m-tiles
    const int mbase = tm * 32, nbase = tn * 32;
    const _Float16* a0 = xh + (size_t)(mbase + n16) * DM;
    const _Float16* a1 = a0 + (size_t)16 * DM;
    const _Float16* b0 = W  + (size_t)(nbase + n16) * DM;
    const _Float16* b1 = b0 + (size_t)16 * DM;
#pragma unroll 2
    for (int kk = 0; kk < DM; kk += 32) {
      v16h fa0 = frag2(a0 + kk + h16 * 16, a0 + kk + h16 * 16 + 8);
      v16h fa1 = frag2(a1 + kk + h16 * 16, a1 + kk + h16 * 16 + 8);
      v16h fb0 = frag2(b0 + kk + h16 * 16, b0 + kk + h16 * 16 + 8);
      v16h fb1 = frag2(b1 + kk + h16 * 16, b1 + kk + h16 * 16 + 8);
      c00 = wmma_f16(fa0, fb0, c00);
      c01 = wmma_f16(fa0, fb1, c01);
      c10 = wmma_f16(fa1, fb0, c10);
      c11 = wmma_f16(fa1, fb1, c11);
    }
    const float sc = (mat == 0) ? 0.125f : 1.0f;   // fold 1/sqrt(64) into Q
    _Float16* dst = (mat == 0) ? Qh : Kh;
    const int h = nbase >> 6;                       // 32 | 64 => single head
    const int b = mbase >> 10, tokb = mbase & 1023;
    const v8f cc[2][2] = {{c00, c01}, {c10, c11}};
#pragma unroll
    for (int i = 0; i < 2; ++i) {
#pragma unroll
      for (int j = 0; j < 2; ++j) {
        const int db = (nbase & 63) + j * 16;
        const float bn = bias[nbase + j * 16 + n16];
#pragma unroll
        for (int r = 0; r < 8; ++r) {
          int tok = tokb + i * 16 + r + 8 * h16;
          dst[(size_t)((b * NH + h) * NTOK + tok) * DH + db + n16] =
              (_Float16)((cc[i][j][r] + bn) * sc);
        }
      }
    }
  } else {
    const int b  = t2 >> 9;                        // 512 tiles per batch
    const int r2 = t2 & 511;
    const int tf = r2 & 15, tt = r2 >> 4;          // 16 feat-tiles x 32 tok-tiles
    const int fbase = tf * 32, tokbase = tt * 32;
    const _Float16* a0 = vwh + (size_t)(fbase + n16) * DM;
    const _Float16* a1 = a0 + (size_t)16 * DM;
    const _Float16* b0 = xh + (size_t)(b * NTOK + tokbase + n16) * DM;
    const _Float16* b1 = b0 + (size_t)16 * DM;
#pragma unroll 2
    for (int kk = 0; kk < DM; kk += 32) {
      v16h fa0 = frag2(a0 + kk + h16 * 16, a0 + kk + h16 * 16 + 8);
      v16h fa1 = frag2(a1 + kk + h16 * 16, a1 + kk + h16 * 16 + 8);
      v16h fb0 = frag2(b0 + kk + h16 * 16, b0 + kk + h16 * 16 + 8);
      v16h fb1 = frag2(b1 + kk + h16 * 16, b1 + kk + h16 * 16 + 8);
      c00 = wmma_f16(fa0, fb0, c00);
      c01 = wmma_f16(fa0, fb1, c01);
      c10 = wmma_f16(fa1, fb0, c10);
      c11 = wmma_f16(fa1, fb1, c11);
    }
    const v8f cc[2][2] = {{c00, c01}, {c10, c11}};
#pragma unroll
    for (int i = 0; i < 2; ++i) {
#pragma unroll
      for (int j = 0; j < 2; ++j) {
#pragma unroll
        for (int r = 0; r < 8; ++r) {
          int f = fbase + i * 16 + r + 8 * h16;
          float v = cc[i][j][r] + vb[f];
          Vth[(size_t)(b * DM + f) * NTOK + tokbase + j * 16 + n16] = (_Float16)v;
        }
      }
    }
  }
}

// ---------------- kernel 2: attention (flash-style, one wave = 16 queries) ---
__global__ void k_attn(const _Float16* __restrict__ Qh, const _Float16* __restrict__ Kh,
                       const _Float16* __restrict__ Vth,
                       const int* __restrict__ positions, const float* __restrict__ mask,
                       const float* __restrict__ dist_emb, const float* __restrict__ order_bias,
                       _Float16* __restrict__ Ah) {
  __shared__ float    sde[NBINS * NH];       // 32x8 distance embedding
  __shared__ _Float16 Pb[4][16 * 40];        // per-wave P tile, 80B row stride (bank-safe)
  for (int i = threadIdx.x; i < NBINS * NH; i += blockDim.x) sde[i] = dist_emb[i];
  __syncthreads();

  const int lane = threadIdx.x & 31;
  const int wv   = threadIdx.x >> 5;
  const int n16  = lane & 15, h16 = lane >> 4;
  const int t  = blockIdx.x * 4 + wv;        // 0..4095
  const int qt = t & 63;
  const int bh = t >> 6;
  const int h = bh & 7, b = bh >> 3;
  const int qbase = qt * 16;

  const _Float16* Qb_ = Qh  + (size_t)((b * NH + h) * NTOK + qbase) * DH;
  const _Float16* Kb_ = Kh  + (size_t)((b * NH + h) * NTOK) * DH;
  const _Float16* Vb_ = Vth + (size_t)(b * DM + h * DH) * NTOK;
  const int*   posb = positions + b * NTOK;
  const float* mb   = mask + b * NTOK;
  const float  obh  = order_bias[h];

  // Q fragments (d 0..31 and 32..63), resident for entire key sweep
  const _Float16* qrow = Qb_ + (size_t)n16 * DH;
  const v16h qf0 = frag2(qrow + h16 * 16,      qrow + h16 * 16 + 8);
  const v16h qf1 = frag2(qrow + 32 + h16 * 16, qrow + 32 + h16 * 16 + 8);

  float pq[8], mq[8];
#pragma unroll
  for (int r = 0; r < 8; ++r) {
    int row = qbase + r + 8 * h16;
    pq[r] = (float)posb[row];
    mq[r] = mb[row];
  }

  float M[8], L[8];
  v8f O0 = {}, O1 = {}, O2 = {}, O3 = {};
#pragma unroll
  for (int r = 0; r < 8; ++r) { M[r] = -1e30f; L[r] = 0.f; }

  _Float16*       prow_st = &Pb[wv][0];
  const _Float16* prow_ld = &Pb[wv][n16 * 40 + h16 * 16];

  for (int kc = 0; kc < NTOK; kc += 32) {
    if (kc + 32 < NTOK)
      __builtin_prefetch(Kb_ + (size_t)(kc + 32 + n16) * DH, 0, 3);

    // S tiles: cols kc..kc+15 (s0), kc+16..kc+31 (s1)
    v8f s0 = {}, s1 = {};
    {
      const _Float16* kr0 = Kb_ + (size_t)(kc + n16) * DH;
      const _Float16* kr1 = Kb_ + (size_t)(kc + 16 + n16) * DH;
      v16h b00 = frag2(kr0 + h16 * 16,      kr0 + h16 * 16 + 8);
      v16h b01 = frag2(kr0 + 32 + h16 * 16, kr0 + 32 + h16 * 16 + 8);
      v16h b10 = frag2(kr1 + h16 * 16,      kr1 + h16 * 16 + 8);
      v16h b11 = frag2(kr1 + 32 + h16 * 16, kr1 + 32 + h16 * 16 + 8);
      s0 = wmma_f16(qf0, b00, s0);
      s0 = wmma_f16(qf1, b01, s0);
      s1 = wmma_f16(qf0, b10, s1);
      s1 = wmma_f16(qf1, b11, s1);
    }
    const float pj0 = (float)posb[kc + n16];
    const float pj1 = (float)posb[kc + 16 + n16];
    const float mk0 = mb[kc + n16];
    const float mk1 = mb[kc + 16 + n16];

    float e0a[8], e1a[8], rm[8];
#pragma unroll
    for (int r = 0; r < 8; ++r) {
      const float pi = pq[r];
      // logspace(0,3,31) edges => searchsorted == ceil(10*log10(dist)), clamped
      float d0 = fabsf(pj0 - pi);
      int bin0 = (int)fminf(31.f, fmaxf(0.f, ceilf(10.f * log10f(d0))));
      float sv0 = s0[r] + sde[bin0 * NH + h] + ((pj0 > pi) ? 0.5f : -0.5f) * obh;
      if (mk0 * mq[r] == 0.f) sv0 = -1e9f;
      float d1 = fabsf(pj1 - pi);
      int bin1 = (int)fminf(31.f, fmaxf(0.f, ceilf(10.f * log10f(d1))));
      float sv1 = s1[r] + sde[bin1 * NH + h] + ((pj1 > pi) ? 0.5f : -0.5f) * obh;
      if (mk1 * mq[r] == 0.f) sv1 = -1e9f;
      e0a[r] = sv0; e1a[r] = sv1;
      float mx = fmaxf(sv0, sv1);           // row max across 16 lanes of half-wave
      mx = fmaxf(mx, __shfl_xor(mx, 1));
      mx = fmaxf(mx, __shfl_xor(mx, 2));
      mx = fmaxf(mx, __shfl_xor(mx, 4));
      mx = fmaxf(mx, __shfl_xor(mx, 8));
      rm[r] = mx;
    }
    float alpha[8];
#pragma unroll
    for (int r = 0; r < 8; ++r) {
      float mn = fmaxf(M[r], rm[r]);
      alpha[r] = __expf(M[r] - mn);
      M[r] = mn;
      float e0 = __expf(e0a[r] - mn);
      float e1 = __expf(e1a[r] - mn);
      e0a[r] = e0; e1a[r] = e1;
      float rs = e0 + e1;
      rs += __shfl_xor(rs, 1); rs += __shfl_xor(rs, 2);
      rs += __shfl_xor(rs, 4); rs += __shfl_xor(rs, 8);
      L[r] = L[r] * alpha[r] + rs;
    }
    // P (16x32 f16) -> LDS in row-major, then reload as A fragment
#pragma unroll
    for (int r = 0; r < 8; ++r) {
      int m = r + 8 * h16;
      prow_st[m * 40 + n16]      = (_Float16)e0a[r];
      prow_st[m * 40 + 16 + n16] = (_Float16)e1a[r];
    }
    asm volatile("s_wait_dscnt 0x0" ::: "memory");
    v16h pa = frag2(prow_ld, prow_ld + 8);

#pragma unroll
    for (int r = 0; r < 8; ++r) {
      O0[r] *= alpha[r]; O1[r] *= alpha[r]; O2[r] *= alpha[r]; O3[r] *= alpha[r];
    }
    // O[dt] += P @ V  (V^T rows give contiguous B-fragment columns)
    {
      const _Float16* vr0 = Vb_ + (size_t)(0 * 16 + n16) * NTOK + kc + h16 * 16;
      const _Float16* vr1 = Vb_ + (size_t)(1 * 16 + n16) * NTOK + kc + h16 * 16;
      const _Float16* vr2 = Vb_ + (size_t)(2 * 16 + n16) * NTOK + kc + h16 * 16;
      const _Float16* vr3 = Vb_ + (size_t)(3 * 16 + n16) * NTOK + kc + h16 * 16;
      O0 = wmma_f16(pa, frag2(vr0, vr0 + 8), O0);
      O1 = wmma_f16(pa, frag2(vr1, vr1 + 8), O1);
      O2 = wmma_f16(pa, frag2(vr2, vr2 + 8), O2);
      O3 = wmma_f16(pa, frag2(vr3, vr3 + 8), O3);
    }
  }

  float inv[8];
#pragma unroll
  for (int r = 0; r < 8; ++r) inv[r] = (L[r] > 0.f) ? (mq[r] / L[r]) : 0.f;
#pragma unroll
  for (int r = 0; r < 8; ++r) {
    int row = qbase + r + 8 * h16;
    _Float16* dst = Ah + (size_t)(b * NTOK + row) * DM + h * DH;
    dst[0 * 16 + n16] = (_Float16)(O0[r] * inv[r]);
    dst[1 * 16 + n16] = (_Float16)(O1[r] * inv[r]);
    dst[2 * 16 + n16] = (_Float16)(O2[r] * inv[r]);
    dst[3 * 16 + n16] = (_Float16)(O3[r] * inv[r]);
  }
}

// ------------- kernel 3: output projection (WMMA, 2x2 reg-blocked), f32 out --
__global__ void k_oproj(const _Float16* __restrict__ Ah, const _Float16* __restrict__ owh,
                        const float* __restrict__ ob, float* __restrict__ out) {
  const int lane = threadIdx.x & 31;
  const int wv   = threadIdx.x >> 5;
  const int t    = blockIdx.x * 4 + wv;   // 0..4095
  const int n16  = lane & 15, h16 = lane >> 4;
  const int tn = t & 15, tm = t >> 4;     // 16 n-tiles x 256 m-tiles
  const int mbase = tm * 32, nbase = tn * 32;
  const _Float16* a0 = Ah  + (size_t)(mbase + n16) * DM;
  const _Float16* a1 = a0 + (size_t)16 * DM;
  const _Float16* b0 = owh + (size_t)(nbase + n16) * DM;
  const _Float16* b1 = b0 + (size_t)16 * DM;
  v8f c00 = {}, c01 = {}, c10 = {}, c11 = {};
#pragma unroll 2
  for (int kk = 0; kk < DM; kk += 32) {
    v16h fa0 = frag2(a0 + kk + h16 * 16, a0 + kk + h16 * 16 + 8);
    v16h fa1 = frag2(a1 + kk + h16 * 16, a1 + kk + h16 * 16 + 8);
    v16h fb0 = frag2(b0 + kk + h16 * 16, b0 + kk + h16 * 16 + 8);
    v16h fb1 = frag2(b1 + kk + h16 * 16, b1 + kk + h16 * 16 + 8);
    c00 = wmma_f16(fa0, fb0, c00);
    c01 = wmma_f16(fa0, fb1, c01);
    c10 = wmma_f16(fa1, fb0, c10);
    c11 = wmma_f16(fa1, fb1, c11);
  }
  const v8f cc[2][2] = {{c00, c01}, {c10, c11}};
#pragma unroll
  for (int i = 0; i < 2; ++i) {
#pragma unroll
    for (int j = 0; j < 2; ++j) {
      const float bn = ob[nbase + j * 16 + n16];
#pragma unroll
      for (int r = 0; r < 8; ++r)
        out[(size_t)(mbase + i * 16 + r + 8 * h16) * DM + nbase + j * 16 + n16] =
            cc[i][j][r] + bn;
    }
  }
}

// ---------------- launch ----------------
extern "C" void kernel_launch(void* const* d_in, const int* in_sizes, int n_in,
                              void* d_out, int out_size, void* d_ws, size_t ws_size,
                              hipStream_t stream) {
  const float* x          = (const float*)d_in[0];
  const int*   positions  = (const int*)  d_in[1];
  const float* mask       = (const float*)d_in[2];
  const float* qw         = (const float*)d_in[3];
  const float* qb         = (const float*)d_in[4];
  const float* kw         = (const float*)d_in[5];
  const float* kb         = (const float*)d_in[6];
  const float* vw         = (const float*)d_in[7];
  const float* vb         = (const float*)d_in[8];
  const float* ow         = (const float*)d_in[9];
  const float* ob         = (const float*)d_in[10];
  const float* dist_emb   = (const float*)d_in[11];
  const float* order_bias = (const float*)d_in[12];
  // d_in[13] bin_edges: replaced by closed-form ceil(10*log10(dist))
  float* out = (float*)d_out;

  const int XE = NB * NTOK * DM;    // 4194304
  const int WE = DM * DM;           // 262144
  _Float16* ws  = (_Float16*)d_ws;
  _Float16* xh  = ws;
  _Float16* qwh = xh  + XE;
  _Float16* kwh = qwh + WE;
  _Float16* vwh = kwh + WE;
  _Float16* owh = vwh + WE;
  _Float16* Qh  = owh + WE;         // (B,H,N,64)
  _Float16* Kh  = Qh  + XE;         // (B,H,N,64)
  _Float16* Vth = Kh  + XE;         // (B,512,1024) = V transposed
  _Float16* Ahb = Vth + XE;         // (B,N,512) attention output (f16)

  k_cvt<<<4096, 256, 0, stream>>>(x,  xh,  XE);
  k_cvt<<<512,  256, 0, stream>>>(qw, qwh, WE);
  k_cvt<<<512,  256, 0, stream>>>(kw, kwh, WE);
  k_cvt<<<512,  256, 0, stream>>>(vw, vwh, WE);
  k_cvt<<<512,  256, 0, stream>>>(ow, owh, WE);

  // 3 * 4096 wave-tiles (32x32 each), 4 waves (128 threads) per block
  k_qkv<<<3072, 128, 0, stream>>>(xh, qwh, qb, kwh, kb, vwh, vb, Qh, Kh, Vth);

  // 64 (b,h) * 64 q-tiles = 4096 waves
  k_attn<<<1024, 128, 0, stream>>>(Qh, Kh, Vth, positions, mask, dist_emb, order_bias, Ahb);

  // 4096 wave-tiles (32x32 each)
  k_oproj<<<1024, 128, 0, stream>>>(Ahb, owh, ob, out);
}
